// emb_sum_layers_34479997452526
// MI455X (gfx1250) — matrix operand order repaired and checked by reading, not compiled, not run on hardware
//
#include <hip/hip_runtime.h>
#include <stdint.h>

// ---------- types ----------
typedef __attribute__((ext_vector_type(16))) __bf16    v16bf;
typedef __attribute__((ext_vector_type(8)))  float     v8f;
typedef __attribute__((ext_vector_type(4)))  unsigned  u32x4;
typedef __attribute__((ext_vector_type(2)))  unsigned  u32x2;

union FragAB { u32x4 q[2]; v16bf v; };

__device__ __forceinline__ unsigned short f32_to_bf16(float f) {
    union { float f; unsigned u; } c; c.f = f;
    unsigned u = c.u;
    u += 0x7FFFu + ((u >> 16) & 1u);   // round-to-nearest-even
    return (unsigned short)(u >> 16);
}
__device__ __forceinline__ float bf16_to_f32(unsigned short h) {
    union { unsigned u; float f; } c; c.u = ((unsigned)h) << 16;
    return c.f;
}

// ---------- small utility kernels ----------
__global__ void fill_zero_f32(float* __restrict__ p, long long n) {
    long long i = (long long)blockIdx.x * blockDim.x + threadIdx.x;
    if (i < n) p[i] = 0.0f;
}

__global__ void deg_count(const int* __restrict__ ei, const int* __restrict__ et,
                          float* __restrict__ deg, int E, int R) {
    int e = blockIdx.x * blockDim.x + threadIdx.x;
    if (e >= E) return;
    int dst = ei[E + e];
    int r   = et[e];
    atomicAdd(deg + (size_t)dst * R + r, 1.0f);
}

__global__ void invert_deg(float* __restrict__ d, long long n) {
    long long i = (long long)blockIdx.x * blockDim.x + threadIdx.x;
    if (i >= n) return;
    float v = d[i];
    d[i] = (v > 0.0f) ? (1.0f / v) : 0.0f;
}

__global__ void cvt_f32_to_bf16(const float* __restrict__ src,
                                unsigned short* __restrict__ dst, long long n) {
    long long i = (long long)blockIdx.x * blockDim.x + threadIdx.x;
    if (i < n) dst[i] = f32_to_bf16(src[i]);
}

// src: fp32 [R][K][Nout] row-major  ->  dst: bf16 [R][Nout][K]  (per-relation transpose)
__global__ void cvt_transpose(const float* __restrict__ src, unsigned short* __restrict__ dst,
                              int R, int K, int Nout) {
    long long i   = (long long)blockIdx.x * blockDim.x + threadIdx.x;
    long long tot = (long long)R * K * Nout;
    if (i >= tot) return;
    int per = K * Nout;
    int r   = (int)(i / per);
    int rem = (int)(i % per);
    int k   = rem / Nout;
    int nn  = rem % Nout;
    dst[(size_t)r * per + (size_t)nn * K + k] = f32_to_bf16(src[i]);
}

__global__ void relu_cvt(const float* __restrict__ src, unsigned short* __restrict__ dst,
                         long long n) {
    long long i = (long long)blockIdx.x * blockDim.x + threadIdx.x;
    if (i < n) dst[i] = f32_to_bf16(fmaxf(src[i], 0.0f));
}

__global__ void sigmoid_inplace(float* __restrict__ p, long long n) {
    long long i = (long long)blockIdx.x * blockDim.x + threadIdx.x;
    if (i < n) p[i] = 1.0f / (1.0f + __expf(-p[i]));
}

// ---------- WMMA GEMM:  C[M,NOUT] = A(bf16,[M,K]) @ B(bf16,[K,NOUT]) (+bias) ----------
// BT is pre-transposed [NOUT,K]; blockIdx.y selects relation slice of BT and C.
// One wave per 16-row stripe; NT = NOUT/16 column tiles; K-loop in 32-chunks of
// v_wmma_f32_16x16x32_bf16.  Fragment layouts per CDNA5 ISA 7.12.2.
template<int NOUT, bool OUT_BF16, bool HAS_BIAS>
__global__ __launch_bounds__(256)
void wmma_gemm_bf16(const unsigned short* __restrict__ A,
                    const unsigned short* __restrict__ BT,
                    const float* __restrict__ bias,
                    void* __restrict__ Cbase,
                    int M, int K, size_t relBStride, size_t relCStride) {
    constexpr int NT = NOUT / 16;
    const int lane    = threadIdx.x & 31;
    const int wave    = threadIdx.x >> 5;
    const int rowTile = blockIdx.x * 8 + wave;
    if (rowTile * 16 >= M) return;

    const unsigned short* Bt = BT + (size_t)blockIdx.y * relBStride;

    v8f acc[NT];
#pragma unroll
    for (int t = 0; t < NT; ++t)
#pragma unroll
        for (int j = 0; j < 8; ++j) acc[t][j] = 0.0f;

    // A-fragment addressing: lane<16 -> M=lane, K pairs {0..7,16..23}; lane>=16 -> {8..15,24..31}
    const int m     = rowTile * 16 + (lane & 15);
    const int abase = (lane < 16) ? 0 : 8;
    // B-fragment addressing: lane%16 = column, lanes 0-15 hold K {0..7,16..23}, lanes 16-31 hold K {8..15,24..31}
    const int n     = lane & 15;
    const int bhalf = (lane < 16) ? 0 : 8;

    const unsigned short* Arow = A + (size_t)m * K;

    for (int kc = 0; kc < K; kc += 32) {
        FragAB af;
        af.q[0] = *reinterpret_cast<const u32x4*>(Arow + kc + abase);        // K+0..7
        af.q[1] = *reinterpret_cast<const u32x4*>(Arow + kc + abase + 16);   // K+16..23
#pragma unroll
        for (int nt = 0; nt < NT; ++nt) {
            FragAB bf_;
            const unsigned short* Bcol = Bt + (size_t)(nt * 16 + n) * K + kc + bhalf;
            bf_.q[0] = *reinterpret_cast<const u32x4*>(Bcol);
            bf_.q[1] = *reinterpret_cast<const u32x4*>(Bcol + 16);
            acc[nt] = __builtin_amdgcn_wmma_f32_16x16x32_bf16(
                false, af.v, false, bf_.v, (short)0, acc[nt], false, false);
        }
    }

    // C/D layout: VGPR j holds row M=j (lanes 0-15) or M=8+j (lanes 16-31), col = lane%16
    const int rbase = rowTile * 16 + ((lane < 16) ? 0 : 8);
#pragma unroll
    for (int nt = 0; nt < NT; ++nt) {
        const int col = nt * 16 + n;
        float bv = 0.0f;
        if (HAS_BIAS) bv = bias[col];
#pragma unroll
        for (int j = 0; j < 8; ++j) {
            float v = acc[nt][j] + bv;
            if (OUT_BF16) {
                unsigned short* C = (unsigned short*)Cbase + (size_t)blockIdx.y * relCStride;
                C[(size_t)(rbase + j) * NOUT + col] = f32_to_bf16(v);
            } else {
                float* C = (float*)Cbase;
                C[(size_t)(rbase + j) * NOUT + col] = v;
            }
        }
    }
}

// ---------- edge gather / mean-normalized scatter-add ----------
// D/4 threads per edge; each thread moves 4 bf16 message elems and does 4 fp32 atomics.
template<int D>
__global__ __launch_bounds__(256)
void edge_scatter(const int* __restrict__ ei, const int* __restrict__ et,
                  const float* __restrict__ inv, const unsigned short* __restrict__ hx,
                  float* __restrict__ agg, int E, int N, int R) {
    constexpr int CH = D / 4;
    long long gid = (long long)blockIdx.x * blockDim.x + threadIdx.x;
    long long e = gid / CH;
    int c = (int)(gid % CH);
    if (e >= E) return;
    int src = ei[e];
    int dst = ei[E + e];
    int r   = et[e];
    float s = inv[(size_t)dst * R + r];
    u32x2 pk = *reinterpret_cast<const u32x2*>(hx + ((size_t)r * N + src) * D + c * 4);
    float* out = agg + (size_t)dst * D + c * 4;
    atomicAdd(out + 0, bf16_to_f32((unsigned short)(pk.x & 0xFFFFu)) * s);
    atomicAdd(out + 1, bf16_to_f32((unsigned short)(pk.x >> 16))     * s);
    atomicAdd(out + 2, bf16_to_f32((unsigned short)(pk.y & 0xFFFFu)) * s);
    atomicAdd(out + 3, bf16_to_f32((unsigned short)(pk.y >> 16))     * s);
}

// ---------- driver ----------
extern "C" void kernel_launch(void* const* d_in, const int* in_sizes, int n_in,
                              void* d_out, int out_size, void* d_ws, size_t ws_size,
                              hipStream_t stream) {
    const int*   edge_index = (const int*)d_in[0];
    const int*   edge_type  = (const int*)d_in[1];
    const float* emb        = (const float*)d_in[2];
    const float* w1         = (const float*)d_in[3];
    const float* root1      = (const float*)d_in[4];
    const float* bias1      = (const float*)d_in[5];
    const float* w2         = (const float*)d_in[6];
    const float* root2      = (const float*)d_in[7];
    const float* bias2      = (const float*)d_in[8];

    const int EMB = 128, HID = 64, LAB = 32;
    const int E = in_sizes[1];
    const int N = in_sizes[2] / EMB;
    const int R = in_sizes[3] / (EMB * HID);

    // carve workspace (256B aligned)
    size_t off = 0;
    auto carve = [&](size_t bytes) -> void* {
        off = (off + 255) & ~(size_t)255;
        void* p = (char*)d_ws + off;
        off += bytes;
        return p;
    };
    float*          invd = (float*)carve((size_t)N * R * 4);                 // deg -> inv, 3.2 MB
    unsigned short* xbf  = (unsigned short*)carve((size_t)N * EMB * 2);      // 25.6 MB
    unsigned short* w1T  = (unsigned short*)carve((size_t)R * EMB * HID * 2);
    unsigned short* r1T  = (unsigned short*)carve((size_t)EMB * HID * 2);
    unsigned short* w2T  = (unsigned short*)carve((size_t)R * HID * LAB * 2);
    unsigned short* r2T  = (unsigned short*)carve((size_t)HID * LAB * 2);
    unsigned short* hx   = (unsigned short*)carve((size_t)R * N * HID * 2);  // 102.4 MB (L2-resident), reused for hx2
    float*          agg1 = (float*)carve((size_t)N * HID * 4);               // 25.6 MB
    unsigned short* hbf  = (unsigned short*)carve((size_t)N * HID * 2);      // 12.8 MB

    const dim3 blk(256);
    const long long nNR = (long long)N * R;

    // degrees -> inverse
    fill_zero_f32<<<(unsigned)((nNR + 255) / 256), blk, 0, stream>>>(invd, nNR);
    deg_count<<<(E + 255) / 256, blk, 0, stream>>>(edge_index, edge_type, invd, E, R);
    invert_deg<<<(unsigned)((nNR + 255) / 256), blk, 0, stream>>>(invd, nNR);

    // operand conversion (weights transposed for contiguous B-fragment loads)
    long long nx = (long long)N * EMB;
    cvt_f32_to_bf16<<<(unsigned)((nx + 255) / 256), blk, 0, stream>>>(emb, xbf, nx);
    cvt_transpose<<<(unsigned)(((long long)R * EMB * HID + 255) / 256), blk, 0, stream>>>(w1, w1T, R, EMB, HID);
    cvt_transpose<<<(unsigned)(((long long)EMB * HID + 255) / 256), blk, 0, stream>>>(root1, r1T, 1, EMB, HID);
    cvt_transpose<<<(unsigned)(((long long)R * HID * LAB + 255) / 256), blk, 0, stream>>>(w2, w2T, R, HID, LAB);
    cvt_transpose<<<(unsigned)(((long long)HID * LAB + 255) / 256), blk, 0, stream>>>(root2, r2T, 1, HID, LAB);

    const int rowTiles = (N + 15) / 16;          // 6250
    dim3 gR((rowTiles + 7) / 8, R);
    dim3 g1((rowTiles + 7) / 8, 1);

    // ---- layer 1 ----
    wmma_gemm_bf16<64, true,  false><<<gR, blk, 0, stream>>>(
        xbf, w1T, nullptr, hx, N, EMB, (size_t)EMB * HID, (size_t)N * HID);
    wmma_gemm_bf16<64, false, true ><<<g1, blk, 0, stream>>>(
        xbf, r1T, bias1, agg1, N, EMB, 0, 0);
    long long t1 = (long long)E * (HID / 4);
    edge_scatter<64><<<(unsigned)((t1 + 255) / 256), blk, 0, stream>>>(
        edge_index, edge_type, invd, hx, agg1, E, N, R);
    long long nh = (long long)N * HID;
    relu_cvt<<<(unsigned)((nh + 255) / 256), blk, 0, stream>>>(agg1, hbf, nh);

    // ---- layer 2 (hx buffer reused for hx2) ----
    wmma_gemm_bf16<32, true,  false><<<gR, blk, 0, stream>>>(
        hbf, w2T, nullptr, hx, N, HID, (size_t)HID * LAB, (size_t)N * LAB);
    wmma_gemm_bf16<32, false, true ><<<g1, blk, 0, stream>>>(
        hbf, r2T, bias2, (float*)d_out, N, HID, 0, 0);
    long long t2 = (long long)E * (LAB / 4);
    edge_scatter<32><<<(unsigned)((t2 + 255) / 256), blk, 0, stream>>>(
        edge_index, edge_type, invd, hx, (float*)d_out, E, N, R);

    long long no = (long long)N * LAB;
    sigmoid_inplace<<<(unsigned)((no + 255) / 256), blk, 0, stream>>>((float*)d_out, no);
}
